// LightshiftaddLayer_171798692039
// MI455X (gfx1250) — matrix operand order repaired and checked by reading, not compiled, not run on hardware
//
#include <hip/hip_runtime.h>
#include <hip/hip_bf16.h>

typedef __bf16 bf16_t;
typedef __attribute__((ext_vector_type(16))) __bf16 v16bf;
typedef __attribute__((ext_vector_type(8)))  float  v8f;

namespace {
constexpr int T      = 4096;
constexpr int BATCH  = 16;
constexpr int C      = 512;
constexpr int H      = 8;
constexpr int KW     = 7;
constexpr int PADL   = 3;

constexpr int TT     = 26;            // output t-values per block
constexpr int TC     = TT + 6;        // computed t-values (with +/-3 halo) = 32
constexpr int MROWS  = TC * BATCH;    // 512 y-rows computed per block
constexpr int NT     = 64;            // output channels per block
constexpr int KC     = 32;            // K chunk (bf16 WMMA depth)
constexpr int NKC    = C / KC;        // 16
constexpr int APITCH = 40;            // bf16 elems per LDS A row (80B: conflict-free frag loads)
constexpr int BPITCH = 40;
constexpr int THREADS = 512;          // 16 wave32
constexpr int MT_PER_WAVE = 2;        // 32 M-tiles / 16 waves
constexpr int NTILES = NT / 16;       // 4
constexpr int XITERS = (MROWS * KC / 8) / THREADS;  // 4 (8 floats per thread per iter)
constexpr int YBYTES = MROWS * NT * 4;       // 131072
constexpr int ABYTES = MROWS * APITCH * 2;   // 40960 hi + 40960 lo + B tile, all alias y
constexpr int SMEM_BYTES = YBYTES + 256;     // + softmax(56 floats)
}

union AFrag { v16bf v; uint4 q[2]; };
union H8    { bf16_t h[8]; uint4 q; };
union H4    { bf16_t h[4]; uint2 q; };

__global__ __launch_bounds__(THREADS)
void lightshift_fused(const float* __restrict__ x, const float* __restrict__ Wsh,
                      const float* __restrict__ bias, const float* __restrict__ wconv,
                      float* __restrict__ out)
{
  extern __shared__ char smem[];
  float*  yS  = (float*)smem;                   // [MROWS][NT] fp32 (aliases A/B below)
  bf16_t* Ahi = (bf16_t*)smem;                  // [MROWS][APITCH]
  bf16_t* Alo = (bf16_t*)(smem + ABYTES);       // [MROWS][APITCH]
  bf16_t* Bs  = (bf16_t*)(smem + 2 * ABYTES);   // [NT][BPITCH]
  float*  wsm = (float*)(smem + YBYTES);        // [H][KW] softmaxed conv weights

  const int tid = threadIdx.x;
  const int nb  = blockIdx.x;                   // channel block (fast dim -> L2 x reuse)
  const int mb  = blockIdx.y;                   // time block
  const int d0  = nb * NT;
  const int t0  = mb * TT;
  const long m0 = (long)(t0 - PADL) * BATCH;    // first (halo) y-row, may be negative

  // softmax of the tiny (8x7) conv kernel, once per block
  if (tid < H) {
    float v[KW]; float mx = -1e30f;
    for (int k = 0; k < KW; ++k) { v[k] = wconv[tid * KW + k]; mx = fmaxf(mx, v[k]); }
    float s = 0.f;
    for (int k = 0; k < KW; ++k) { v[k] = __expf(v[k] - mx); s += v[k]; }
    float inv = 1.f / s;
    for (int k = 0; k < KW; ++k) wsm[tid * KW + k] = v[k] * inv;
  }

  const int wave  = tid >> 5;
  const int lane  = tid & 31;
  const int lhalf = lane >> 4;
  const int l16   = lane & 15;

  // W staging ownership (one float4 per thread per chunk)
  const int wdl  = tid >> 3;                    // 0..63
  const int wcc4 = (tid & 7) * 4;               // 0,4,...,28

  v8f zero; for (int i = 0; i < 8; ++i) zero[i] = 0.f;
  v8f acc[MT_PER_WAVE][NTILES];
  for (int mi = 0; mi < MT_PER_WAVE; ++mi)
    for (int ni = 0; ni < NTILES; ++ni) acc[mi][ni] = zero;

  // ---- software-pipelined staging: chunk k+1 global loads fly under chunk k WMMAs ----
  float  xv[XITERS][8];
  float4 wv;

  auto load_chunk = [&](int kc) {
    #pragma unroll
    for (int it = 0; it < XITERS; ++it) {
      int g  = it * THREADS + tid;
      int r  = g >> 2;              // y-row 0..511
      int cg = (g & 3) * 8;         // 0,8,16,24
      long m = m0 + r;
      if (m >= 0 && m < (long)T * BATCH) {
        const float* p = x + m * C + kc * KC + cg;
        float4 a = *(const float4*)p;
        float4 b = *(const float4*)(p + 4);
        xv[it][0]=a.x; xv[it][1]=a.y; xv[it][2]=a.z; xv[it][3]=a.w;
        xv[it][4]=b.x; xv[it][5]=b.y; xv[it][6]=b.z; xv[it][7]=b.w;
        if (kc + 1 < NKC) __builtin_prefetch(p + KC, 0, 1);  // warm L2 one more chunk out
      } else {
        for (int i = 0; i < 8; ++i) xv[it][i] = 0.f;         // conv zero-padding region
      }
    }
    wv = *(const float4*)(Wsh + (long)(d0 + wdl) * C + kc * KC + wcc4);
  };

  auto store_chunk = [&]() {
    #pragma unroll
    for (int it = 0; it < XITERS; ++it) {
      int g  = it * THREADS + tid;
      int r  = g >> 2;
      int cg = (g & 3) * 8;
      H8 hi, lo;
      #pragma unroll
      for (int i = 0; i < 8; ++i) {
        bf16_t h = (bf16_t)xv[it][i];              // RNE; x = hi + lo (bf16x2 split)
        hi.h[i] = h;
        lo.h[i] = (bf16_t)(xv[it][i] - (float)h);
      }
      *(uint4*)(Ahi + r * APITCH + cg) = hi.q;
      *(uint4*)(Alo + r * APITCH + cg) = lo.q;
    }
    // power-of-2 quantize W (exact in bf16): sign * 2^round(log2|w|+eps)
    float wf[4] = { wv.x, wv.y, wv.z, wv.w };
    H4 b4;
    #pragma unroll
    for (int i = 0; i < 4; ++i) {
      float w  = wf[i];
      float aw = fabsf(w) + 1e-12f;
      int   ei = (int)rintf(__log2f(aw));          // round-half-even like jnp.round
      float q  = (w == 0.f) ? 0.f : copysignf(ldexpf(1.0f, ei), w);
      b4.h[i] = (bf16_t)q;
    }
    *(uint2*)(Bs + wdl * BPITCH + wcc4) = b4.q;
  };

  load_chunk(0);
  for (int kc = 0; kc < NKC; ++kc) {
    __syncthreads();                // all frag reads of previous chunk complete
    store_chunk();                  // regs -> LDS (convert)
    __syncthreads();
    if (kc + 1 < NKC) load_chunk(kc + 1);   // issue HBM loads; consumed next iteration

    // ---- B fragments: lane l16 = column, lane halves take K {0-7,16-23} / {8-15,24-31} ----
    AFrag bfr[NTILES];
    const int k0b = lhalf ? 8 : 0;
    #pragma unroll
    for (int ni = 0; ni < NTILES; ++ni) {
      const bf16_t* bp = Bs + (ni * 16 + l16) * BPITCH + k0b;
      bfr[ni].q[0] = *(const uint4*)bp;
      bfr[ni].q[1] = *(const uint4*)(bp + 16);
    }

    // ---- A fragments + WMMA: lane halves take K {0-7,16-23} / {8-15,24-31} ----
    const int k0a = lhalf ? 8 : 0;
    #pragma unroll
    for (int mi = 0; mi < MT_PER_WAVE; ++mi) {
      int mt = wave * MT_PER_WAVE + mi;
      const bf16_t* ap = Ahi + (mt * 16 + l16) * APITCH + k0a;
      const bf16_t* lp = Alo + (mt * 16 + l16) * APITCH + k0a;
      AFrag ah, al;
      ah.q[0] = *(const uint4*)ap;  ah.q[1] = *(const uint4*)(ap + 16);
      al.q[0] = *(const uint4*)lp;  al.q[1] = *(const uint4*)(lp + 16);
      #pragma unroll
      for (int ni = 0; ni < NTILES; ++ni) {
        acc[mi][ni] = __builtin_amdgcn_wmma_f32_16x16x32_bf16(
            false, ah.v, false, bfr[ni].v, (short)0, acc[mi][ni], false, false);
        acc[mi][ni] = __builtin_amdgcn_wmma_f32_16x16x32_bf16(
            false, al.v, false, bfr[ni].v, (short)0, acc[mi][ni], false, false);
      }
    }
  }

  __syncthreads();  // A/B LDS dead; alias it with fp32 y

  // ---- spill accumulators (+bias) to y LDS; out-of-range t tiles become zeros ----
  #pragma unroll
  for (int mi = 0; mi < MT_PER_WAVE; ++mi) {
    int mt = wave * MT_PER_WAVE + mi;
    int tg = t0 - PADL + mt;                    // each 16-row tile is one t-value
    bool valid = (tg >= 0) && (tg < T);
    #pragma unroll
    for (int ni = 0; ni < NTILES; ++ni) {
      int col = ni * 16 + l16;
      float bv = bias[d0 + col];
      #pragma unroll
      for (int v = 0; v < 8; ++v) {
        int mrow = mt * 16 + v + 8 * lhalf;     // C/D layout: VGPR v -> row v / v+8
        yS[mrow * NT + col] = valid ? (acc[mi][ni][v] + bv) : 0.f;
      }
    }
  }
  __syncthreads();

  // ---- depthwise head-shared conv over LDS y, write final output ----
  const int nout = TT * BATCH * NT;             // 26624 = 52 per thread
  for (int o = tid; o < nout; o += THREADS) {
    int col = o & (NT - 1);
    int rr  = o >> 6;
    int tl  = rr >> 4;
    int bb  = rr & (BATCH - 1);
    int tg  = t0 + tl;
    if (tg >= T) continue;                      // ragged last time-block
    int head = (d0 + col) >> 6;                 // R = C/H = 64
    const float* wk = wsm + head * KW;
    float s = 0.f;
    #pragma unroll
    for (int k = 0; k < KW; ++k) {
      int tin = tg + k - PADL;
      float yv = (tin >= 0 && tin < T) ? yS[((tl + k) * BATCH + bb) * NT + col] : 0.f;
      s += wk[k] * yv;
    }
    out[((long)tg * BATCH + bb) * C + d0 + col] = s;
  }
}

extern "C" void kernel_launch(void* const* d_in, const int* in_sizes, int n_in,
                              void* d_out, int out_size, void* d_ws, size_t ws_size,
                              hipStream_t stream) {
  (void)in_sizes; (void)n_in; (void)d_ws; (void)ws_size; (void)out_size;
  const float* x    = (const float*)d_in[0];   // (T,B,C)
  const float* Wsh  = (const float*)d_in[1];   // (C,C)
  const float* bias = (const float*)d_in[2];   // (C,)
  const float* wcv  = (const float*)d_in[3];   // (H,KW)
  float* out = (float*)d_out;

  dim3 grid(C / NT, (T + TT - 1) / TT);        // nb fastest -> x tile L2 reuse
  lightshift_fused<<<grid, THREADS, SMEM_BYTES, stream>>>(x, Wsh, bias, wcv, out);
}
